// AttnBlock_43267500540469
// MI455X (gfx1250) — compile-verified, matrix operand order checked by reading, not run on hardware
//
#include <hip/hip_runtime.h>
#include <hip/hip_bf16.h>

#define DEVINL __device__ __forceinline__

typedef __attribute__((ext_vector_type(16))) __bf16 bf16x16;
typedef __attribute__((ext_vector_type(8)))  __bf16 bf16x8;
typedef __attribute__((ext_vector_type(8)))  float  f32x8;

constexpr int BN = 4, C = 256, T = 4096;
constexpr float EPS = 1e-5f;
constexpr float NEGV = -1e8f;

DEVINL f32x8 zero8() {
  f32x8 z;
#pragma unroll
  for (int i = 0; i < 8; ++i) z[i] = 0.0f;
  return z;
}

// WMMA bf16 fragment loader for row-major [rows x ld] bf16 data.
// Works for A (row index = M row) and for B when the source is stored
// K-contiguous per output column (i.e. B^T row-major), matching the CDNA5
// 16-bit operand layout: lane L holds index (L&15); K halves chosen by L>>4:
// K in [half*8, half*8+8) and [16+half*8, 16+half*8+8).
DEVINL bf16x16 load_frag(const __bf16* __restrict__ base, int row, int ld,
                         int k0, int lane) {
  int m = lane & 15;
  int half = lane >> 4;
  const __bf16* p = base + (size_t)(row + m) * ld + k0 + half * 8;
  bf16x8 lo = *reinterpret_cast<const bf16x8*>(p);
  bf16x8 hi = *reinterpret_cast<const bf16x8*>(p + 16);
  return __builtin_shufflevector(lo, hi, 0, 1, 2, 3, 4, 5, 6, 7, 8, 9, 10, 11,
                                 12, 13, 14, 15);
}

DEVINL f32x8 wmma_bf16(bf16x16 a, bf16x16 b, f32x8 c) {
  return __builtin_amdgcn_wmma_f32_16x16x32_bf16(false, a, false, b, (short)0,
                                                 c, false, false);
}

// ---------------------------------------------------------------- weights cvt
__global__ void cvt_weights(const float* __restrict__ wp,
                            const float* __restrict__ wq,
                            const float* __restrict__ wk,
                            const float* __restrict__ wv,
                            const float* __restrict__ wo,
                            __bf16* __restrict__ dst) {
  int i = blockIdx.x * blockDim.x + threadIdx.x;
  if (i >= 5 * C * C) return;
  int m = i >> 16;  // C*C == 65536
  int r = i & (C * C - 1);
  const float* src = (m == 0) ? wp : (m == 1) ? wq : (m == 2) ? wk
                     : (m == 3) ? wv : wo;
  dst[i] = (__bf16)src[r];
}

// ------------------------------------------------- mask + transpose + LN
// x [B,C,T] f32 -> ln [B,T,C] bf16 (masked, layernorm over C, gamma/beta)
__global__ __launch_bounds__(256) void ln_kernel(const float* __restrict__ x,
                                                 const float* __restrict__ mask,
                                                 const float* __restrict__ gamma,
                                                 const float* __restrict__ beta,
                                                 __bf16* __restrict__ ln) {
  __shared__ float xs[32][C + 1];
  __shared__ float ps[32][8], ps2[32][8], mu[32], rs[32];
  int b = blockIdx.y;
  int t0 = blockIdx.x * 32;
  int tid = threadIdx.x;
  // coalesced load along t, transpose into LDS as [t][c]
  for (int idx = tid; idx < 32 * C; idx += 256) {
    int tl = idx & 31, c = idx >> 5;
    float v = x[((size_t)b * C + c) * T + t0 + tl] *
              mask[(size_t)b * T + t0 + tl];
    xs[tl][c] = v;
  }
  __syncthreads();
  {
    int tl = tid & 31, seg = tid >> 5;
    float s = 0.f, s2 = 0.f;
    for (int c = seg * 32; c < seg * 32 + 32; ++c) {
      float v = xs[tl][c];
      s += v;
      s2 += v * v;
    }
    ps[tl][seg] = s;
    ps2[tl][seg] = s2;
  }
  __syncthreads();
  if (tid < 32) {
    float s = 0.f, s2 = 0.f;
    for (int seg = 0; seg < 8; ++seg) {
      s += ps[tid][seg];
      s2 += ps2[tid][seg];
    }
    float mean = s * (1.0f / C);
    float var = s2 * (1.0f / C) - mean * mean;
    mu[tid] = mean;
    rs[tid] = rsqrtf(var + EPS);
  }
  __syncthreads();
  // coalesced store along c
  for (int idx = tid; idx < 32 * C; idx += 256) {
    int c = idx & (C - 1), tl = idx >> 8;
    float v = (xs[tl][c] - mu[tl]) * rs[tl] * gamma[c] + beta[c];
    ln[((size_t)b * T + t0 + tl) * C + c] = (__bf16)v;
  }
}

// ----------------------------------------------------------------- GEMM
// out[t,o] = sum_c in[b,t,c] * W[o,c] + bias[o]
// Weight tile (64 x 256) staged once in LDS (padded stride 264 to spread the
// 16 fragment lanes over distinct banks); A streams from global.
// MODE 0: out bf16 [B,T,C], val = (acc+bias)*mask[t]
// MODE 1: out bf16 [B,C,T] (transposed via LDS), val = (acc+bias)*mask[t]
// MODE 2: out f32 [B,C,T], val = (x + (acc+bias)*mask[t]) * mask[t]
template <int MODE>
__global__ __launch_bounds__(128) void gemm_kernel(
    const __bf16* __restrict__ in, const __bf16* __restrict__ W,
    const float* __restrict__ bias, const float* __restrict__ mask,
    const float* __restrict__ xres, __bf16* __restrict__ outb,
    float* __restrict__ outf) {
  constexpr int LDW = C + 8;  // 264 elements, 528B row stride (16B multiple)
  __shared__ __bf16 Wl[64 * LDW];
  __shared__ float sD[64][65];
  int b = blockIdx.z;
  int t0 = blockIdx.x * 64, o0 = blockIdx.y * 64;
  int lane = threadIdx.x & 31, wv = threadIdx.x >> 5;

  // cooperative stage of the 64x256 weight tile into LDS (b128 copies)
  for (int idx = threadIdx.x; idx < 64 * 32; idx += 128) {
    int o = idx >> 5;
    int kk = (idx & 31) * 8;
    bf16x8 v = *reinterpret_cast<const bf16x8*>(W + (size_t)(o0 + o) * C + kk);
    *reinterpret_cast<bf16x8*>(&Wl[o * LDW + kk]) = v;
  }
  __syncthreads();

  const __bf16* inb = in + (size_t)b * T * C;
  f32x8 acc[4];
#pragma unroll
  for (int j = 0; j < 4; ++j) acc[j] = zero8();
  int trow = t0 + wv * 16;
#pragma unroll
  for (int k0 = 0; k0 < C; k0 += 32) {
    bf16x16 a = load_frag(inb, trow, C, k0, lane);
#pragma unroll
    for (int j = 0; j < 4; ++j) {
      bf16x16 bf = load_frag(Wl, j * 16, LDW, k0, lane);
      acc[j] = wmma_bf16(a, bf, acc[j]);
    }
  }
  int half = lane >> 4;
  if (MODE == 0) {
#pragma unroll
    for (int j = 0; j < 4; ++j) {
      int o = o0 + j * 16 + (lane & 15);
      float bo = bias[o];
#pragma unroll
      for (int r = 0; r < 8; ++r) {
        int t = trow + r + 8 * half;
        float mv = mask[(size_t)b * T + t];
        outb[((size_t)b * T + t) * C + o] = (__bf16)((acc[j][r] + bo) * mv);
      }
    }
  } else {
#pragma unroll
    for (int j = 0; j < 4; ++j)
#pragma unroll
      for (int r = 0; r < 8; ++r)
        sD[j * 16 + (lane & 15)][wv * 16 + r + 8 * half] = acc[j][r];
    __syncthreads();
    for (int idx = threadIdx.x; idx < 64 * 64; idx += 128) {
      int lt = idx & 63, lo = idx >> 6;
      int t = t0 + lt, o = o0 + lo;
      float mv = mask[(size_t)b * T + t];
      float v = (sD[lo][lt] + bias[o]) * mv;
      size_t oidx = ((size_t)b * C + o) * T + t;
      if (MODE == 1) {
        outb[oidx] = (__bf16)v;
      } else {
        outf[oidx] = (xres[oidx] + v) * mv;
      }
    }
  }
}

// --------------------------------------------------------- flash attention
// q,k [B,T,C] bf16 ; vT [B,C,T] bf16 ; out attn [B,T,C] bf16
// 32 queries per block, 8 waves. Key blocks of 64, online softmax with
// additive key mask (1-m)*NEG. Wave w: qi = w>>2 selects the 16-query tile,
// sub = w&3 selects the key sub-tile (phase 1) / 64-channel slice (phase 2).
__global__ __launch_bounds__(256) void flash_kernel(
    const __bf16* __restrict__ q, const __bf16* __restrict__ k,
    const __bf16* __restrict__ vT, const float* __restrict__ mask,
    __bf16* __restrict__ attn) {
  __shared__ float S[32][64];
  __shared__ __bf16 P[32][64];
  __shared__ float Mrow[32], Lrow[32], Arow[32], negk[64];
  int b = blockIdx.y;
  int q0 = blockIdx.x * 32;
  int tid = threadIdx.x;
  int lane = tid & 31, wv = tid >> 5;
  int qi = wv >> 2, sub = wv & 3;
  int half = lane >> 4;
  const __bf16* qb = q + (size_t)b * T * C;
  const __bf16* kb = k + (size_t)b * T * C;
  const __bf16* vb = vT + (size_t)b * C * T;
  const float scale = 0.0625f;  // 1/sqrt(256)

  // hoist this wave's q fragments (16 rows x 256 K) into registers
  bf16x16 qf[8];
#pragma unroll
  for (int kk = 0; kk < 8; ++kk)
    qf[kk] = load_frag(qb, q0 + qi * 16, C, kk * 32, lane);

  f32x8 O[4];
#pragma unroll
  for (int j = 0; j < 4; ++j) O[j] = zero8();
  if (tid < 32) {
    Mrow[tid] = -3.0e38f;
    Lrow[tid] = 0.f;
  }
  __syncthreads();

  for (int s0 = 0; s0 < T; s0 += 64) {
    if (tid < 64) {
      float mv = mask[(size_t)b * T + s0 + tid];
      negk[tid] = (1.0f - mv) * NEGV;
    }
    // prefetch next key/value block into cache while we compute
    if (s0 + 64 < T) {
      const __bf16* pk =
          kb + (size_t)(s0 + 64 + sub * 16 + (lane & 15)) * C + half * 128;
      __builtin_prefetch(pk, 0, 3);
      const __bf16* pv =
          vb + (size_t)(sub * 64 + (lane & 15)) * T + s0 + 64 + half * 32;
      __builtin_prefetch(pv, 0, 3);
    }
    // phase 1: each wave computes one 16q x 16s score tile (full C)
    f32x8 sa = zero8();
#pragma unroll
    for (int kk = 0; kk < 8; ++kk) {
      bf16x16 bf = load_frag(kb, s0 + sub * 16, C, kk * 32, lane);
      sa = wmma_bf16(qf[kk], bf, sa);
    }
    __syncthreads();  // prev-iter P consumed; negk visible
    {
      int n = sub * 16 + (lane & 15);
#pragma unroll
      for (int r = 0; r < 8; ++r) S[qi * 16 + r + 8 * half][n] = sa[r];
    }
    __syncthreads();
    // online softmax, one thread per query row
    if (tid < 32) {
      float mOld = Mrow[tid];
      float rmax = -3.0e38f;
      for (int j2 = 0; j2 < 64; ++j2) {
        float v = S[tid][j2] * scale + negk[j2];
        S[tid][j2] = v;
        rmax = fmaxf(rmax, v);
      }
      float mNew = fmaxf(mOld, rmax);
      float alpha = __expf(mOld - mNew);
      float lsum = 0.f;
      for (int j2 = 0; j2 < 64; ++j2) {
        float p = __expf(S[tid][j2] - mNew);
        lsum += p;
        P[tid][j2] = (__bf16)p;
      }
      Mrow[tid] = mNew;
      Lrow[tid] = Lrow[tid] * alpha + lsum;
      Arow[tid] = alpha;
    }
    __syncthreads();
    // phase 2: O[16q x 64c per wave] = O*alpha + P @ V
    float al[8];
#pragma unroll
    for (int r = 0; r < 8; ++r) al[r] = Arow[qi * 16 + r + 8 * half];
#pragma unroll
    for (int j = 0; j < 4; ++j)
#pragma unroll
      for (int r = 0; r < 8; ++r) O[j][r] *= al[r];
    const __bf16* Pp = &P[qi * 16][0];
    bf16x16 pa0 = load_frag(Pp, 0, 64, 0, lane);
    bf16x16 pa1 = load_frag(Pp, 0, 64, 32, lane);
#pragma unroll
    for (int j = 0; j < 4; ++j) {
      int c = sub * 64 + j * 16;
      bf16x16 vb0 = load_frag(vb, c, T, s0, lane);
      bf16x16 vb1 = load_frag(vb, c, T, s0 + 32, lane);
      O[j] = wmma_bf16(pa0, vb0, O[j]);
      O[j] = wmma_bf16(pa1, vb1, O[j]);
    }
  }
  __syncthreads();
  float linv[8];
#pragma unroll
  for (int r = 0; r < 8; ++r) {
    float l = Lrow[qi * 16 + r + 8 * half];
    linv[r] = (l > 0.f) ? 1.0f / l : 0.f;
  }
#pragma unroll
  for (int j = 0; j < 4; ++j) {
    int c = sub * 64 + j * 16 + (lane & 15);
#pragma unroll
    for (int r = 0; r < 8; ++r) {
      int t = q0 + qi * 16 + r + 8 * half;
      attn[((size_t)b * T + t) * C + c] = (__bf16)(O[j][r] * linv[r]);
    }
  }
}

// ---------------------------------------------------------------- launcher
extern "C" void kernel_launch(void* const* d_in, const int* in_sizes, int n_in,
                              void* d_out, int out_size, void* d_ws,
                              size_t ws_size, hipStream_t stream) {
  (void)in_sizes; (void)n_in; (void)out_size; (void)ws_size;
  const float* x     = (const float*)d_in[0];
  const float* xmask = (const float*)d_in[1];
  const float* gamma = (const float*)d_in[2];
  const float* beta  = (const float*)d_in[3];
  const float* Wp    = (const float*)d_in[4];
  const float* bp    = (const float*)d_in[5];
  const float* Wq    = (const float*)d_in[6];
  const float* bq    = (const float*)d_in[7];
  const float* Wk    = (const float*)d_in[8];
  const float* bk    = (const float*)d_in[9];
  const float* Wv    = (const float*)d_in[10];
  const float* bv    = (const float*)d_in[11];
  const float* Wo    = (const float*)d_in[12];
  const float* bo    = (const float*)d_in[13];

  constexpr size_t SZ = (size_t)BN * T * C * 2;  // 8 MB per bf16 tensor
  char* ws = (char*)d_ws;
  __bf16* wb = (__bf16*)(ws);                    // 5 * C*C bf16 = 640 KB
  __bf16* ln = (__bf16*)(ws + 655360);
  __bf16* h  = (__bf16*)(ws + 655360 + 1 * SZ);
  __bf16* qb = (__bf16*)(ws + 655360 + 2 * SZ);
  __bf16* kb = (__bf16*)(ws + 655360 + 3 * SZ);
  __bf16* vT = (__bf16*)(ws + 655360 + 4 * SZ);
  __bf16* at = (__bf16*)(ws + 655360 + 5 * SZ);

  cvt_weights<<<dim3((5 * C * C + 255) / 256), 256, 0, stream>>>(Wp, Wq, Wk,
                                                                 Wv, Wo, wb);
  ln_kernel<<<dim3(T / 32, BN), 256, 0, stream>>>(x, xmask, gamma, beta, ln);

  dim3 gg(T / 64, C / 64, BN);
  gemm_kernel<0><<<gg, 128, 0, stream>>>(ln, wb + 0 * C * C, bp, xmask,
                                         nullptr, h, nullptr);
  gemm_kernel<0><<<gg, 128, 0, stream>>>(h, wb + 1 * C * C, bq, xmask,
                                         nullptr, qb, nullptr);
  gemm_kernel<0><<<gg, 128, 0, stream>>>(h, wb + 2 * C * C, bk, xmask,
                                         nullptr, kb, nullptr);
  gemm_kernel<1><<<gg, 128, 0, stream>>>(h, wb + 3 * C * C, bv, xmask,
                                         nullptr, vT, nullptr);

  flash_kernel<<<dim3(T / 32, BN), 256, 0, stream>>>(qb, kb, vT, xmask, at);

  gemm_kernel<2><<<gg, 128, 0, stream>>>(at, wb + 4 * C * C, bo, xmask, x,
                                         nullptr, (float*)d_out);
}
